// EMA_57011395887305
// MI455X (gfx1250) — matrix operand order repaired and checked by reading, not compiled, not run on hardware
//
#include <hip/hip_runtime.h>

// ---------------------------------------------------------------------------
// EMA block on gfx1250: all GEMMs via v_wmma_f32_16x16x32_bf16 (bf16 in,
// fp32 accumulate). conv0 (0.174 TFLOP) dominates -> implicit-GEMM WMMA conv
// with branchless zero-padding (address select) + software-pipelined k-loop.
// ---------------------------------------------------------------------------

typedef __attribute__((ext_vector_type(16))) __bf16 v16bf;
typedef __attribute__((ext_vector_type(8)))  float  v8f;

#define B_   4
#define CIN  1024
#define CR   256
#define KK   64
#define HH   96
#define WW   96
#define NN   9216          /* H*W */
#define EPS_BN 1e-5f
#define EPS_EM 1e-6f

static __device__ __forceinline__ unsigned short f32_bf16(float f) {
  unsigned x = __float_as_uint(f);
  unsigned r = x + 0x7FFFu + ((x >> 16) & 1u);   // round-to-nearest-even
  return (unsigned short)(r >> 16);
}

static __device__ __forceinline__ unsigned pack_bf16x2(float lo, float hi) {
  return (unsigned)f32_bf16(lo) | ((unsigned)f32_bf16(hi) << 16);
}

// A-fragment: 16x32 bf16 (MxK), row-major [M][K], p at tile origin, ld even.
// lanes0-15 hold K=0..15, lanes16-31 hold K=16..31 (matches B-fragment order).
static __device__ __forceinline__ v16bf load_a_frag(const unsigned short* p,
                                                    int ld, int lane) {
  const int m = lane & 15, hl = lane >> 4;
  const unsigned short* row = p + (size_t)m * ld + hl * 16;
  union { v16bf v; unsigned u[8]; } f;
#pragma unroll
  for (int i = 0; i < 8; ++i) f.u[i] = *(const unsigned*)(row + 2 * i);
  return f.v;
}

// B-fragment: 32x16 bf16 (KxN), stored transposed [N][K] (K contiguous).
// lanes0-15 hold K=0..15, lanes16-31 hold K=16..31 (2 per VGPR).
static __device__ __forceinline__ v16bf load_b_frag(const unsigned short* p,
                                                    int ld, int lane) {
  const int n = lane & 15, hl = lane >> 4;
  const unsigned short* row = p + (size_t)n * ld + hl * 16;
  union { v16bf v; unsigned u[8]; } f;
#pragma unroll
  for (int i = 0; i < 8; ++i) f.u[i] = *(const unsigned*)(row + 2 * i);
  return f.v;
}

// B-fragment where the lane's (possibly zero-page) row pointer is precomputed.
// Unconditional contiguous 32B load -> vectorizes to global_load_b128 x2.
static __device__ __forceinline__ v16bf load_b_row(const unsigned short* rowp,
                                                   int lane) {
  const unsigned short* p = rowp + (lane >> 4) * 16;
  union { v16bf v; unsigned u[8]; } f;
#pragma unroll
  for (int i = 0; i < 8; ++i) f.u[i] = *(const unsigned*)(p + 2 * i);
  return f.v;
}

static __device__ __forceinline__ v8f wmma_bf16(v16bf a, v16bf b, v8f c) {
  return __builtin_amdgcn_wmma_f32_16x16x32_bf16(false, a, false, b,
                                                 (short)0, c, false, false);
}

// ------------------------------- prep kernels ------------------------------

__global__ void prep_bn_k(const float* __restrict__ g0, const float* __restrict__ be0,
                          const float* __restrict__ m0, const float* __restrict__ v0,
                          const float* __restrict__ g2, const float* __restrict__ be2,
                          const float* __restrict__ m2, const float* __restrict__ v2,
                          float* __restrict__ s0, float* __restrict__ t0,
                          float* __restrict__ s2, float* __restrict__ t2,
                          unsigned short* __restrict__ zpad) {
  const int c = threadIdx.x;
  const float i0 = g0[c] * rsqrtf(v0[c] + EPS_BN);
  s0[c] = i0; t0[c] = be0[c] - m0[c] * i0;
  const float i2 = g2[c] * rsqrtf(v2[c] + EPS_BN);
  s2[c] = i2; t2[c] = be2[c] - m2[c] * i2;
  for (int i = c; i < CIN + 64; i += 256) zpad[i] = 0;   // zero page for padding
}

// w0 [co][ci][3][3] fp32 -> [co][tap][ci] bf16 (K-contiguous per output row)
__global__ void pack_w0_k(const float* __restrict__ w0, unsigned short* __restrict__ w0p) {
  const int o = blockIdx.x * 256 + threadIdx.x;       // CR*9*CIN exact
  const int co = o / (9 * CIN);
  const int r  = o % (9 * CIN);
  const int tap = r / CIN, ci = r % CIN;
  w0p[o] = f32_bf16(w0[((size_t)co * CIN + ci) * 9 + tap]);
}

__global__ void pack_bf16_k(const float* __restrict__ src, unsigned short* __restrict__ dst,
                            int n) {
  const int o = blockIdx.x * 256 + threadIdx.x;
  if (o < n) dst[o] = f32_bf16(src[o]);
}

// x [b][ci][n] fp32 -> [b][n][ci] bf16 via LDS tile (coalesced both sides)
__global__ void __launch_bounds__(256)
x_transpose_k(const float* __restrict__ x, unsigned short* __restrict__ xnc) {
  __shared__ unsigned short t[64][65];
  const int b = blockIdx.z;
  const int n0 = blockIdx.x * 64;
  const int c0 = blockIdx.y * 64;
  for (int e = threadIdx.x; e < 64 * 64; e += 256) {
    const int ci = e >> 6, nl = e & 63;     // reads contiguous in n
    t[nl][ci] = f32_bf16(x[((size_t)b * CIN + c0 + ci) * NN + n0 + nl]);
  }
  __syncthreads();
  for (int e = threadIdx.x; e < 64 * 64; e += 256) {
    const int nl = e >> 6, ci = e & 63;     // writes contiguous in ci
    xnc[((size_t)b * NN + n0 + nl) * CIN + c0 + ci] = t[nl][ci];
  }
}

// mu [1][c][k] fp32 -> per-batch bf16 copies in [k][c] and [c][k] layouts
__global__ void mu_init_k(const float* __restrict__ mu_in,
                          unsigned short* __restrict__ mu_kc,
                          unsigned short* __restrict__ mu_ck) {
  const int o = blockIdx.x * 256 + threadIdx.x;       // B*CR*KK exact
  const int b = o >> 14, r = o & 16383;
  const int c = r >> 6, k = r & 63;
  const unsigned short v = f32_bf16(mu_in[c * KK + k]);
  mu_kc[((size_t)b * KK + k) * CR + c] = v;
  mu_ck[((size_t)b * CR + c) * KK + k] = v;
}

__global__ void zero_s_mu_k(float* __restrict__ S, float* __restrict__ mu_raw) {
  const int o = blockIdx.x * 256 + threadIdx.x;
  if (o < B_ * CR * KK) mu_raw[o] = 0.f;
  if (o < B_ * KK) S[o] = 0.f;
}

// ------------------------- conv0: 3x3 implicit GEMM ------------------------
// grid (72, 4, B), block 256. WG tile 64(co) x 128(pix); 8 waves of 2x2 frags.
// Padding handled by per-lane address select onto a zeroed scratch row.
__global__ void __launch_bounds__(256)
conv0_bn_relu_k(const unsigned short* __restrict__ xnc,   // [B][N][CIN]
                const unsigned short* __restrict__ w0p,   // [CR][9][CIN]
                const unsigned short* __restrict__ zpad,  // >= CIN+64 zeros
                const float* __restrict__ s0, const float* __restrict__ t0,
                float* __restrict__ idn,                  // [B][CR][N] fp32
                unsigned short* __restrict__ idn_nc) {    // [B][N][CR] bf16
  const int b = blockIdx.z;
  const int co0 = blockIdx.y * 64;
  const int n0  = blockIdx.x * 128;
  const int lane = threadIdx.x & 31, wave = threadIdx.x >> 5;
  const int com = co0 + (wave >> 2) * 32;
  const int nnb = n0 + (wave & 3) * 32;
  const int ln = lane & 15, hl = lane >> 4;

  v8f acc[2][2] = {};
  const unsigned short* xb = xnc + (size_t)b * NN * CIN;

  const int nA = nnb + ln, nB = nA + 16;
  const int hA = nA / WW, wA = nA % WW;
  const int hB = nB / WW, wB = nB % WW;

#pragma unroll
  for (int tap = 0; tap < 9; ++tap) {
    const int dh = tap / 3 - 1, dw = tap % 3 - 1;
    const bool vA = ((unsigned)(hA + dh) < HH) && ((unsigned)(wA + dw) < WW);
    const bool vB = ((unsigned)(hB + dh) < HH) && ((unsigned)(wB + dw) < WW);
    // branchless: invalid lanes stream zeros from zpad
    const unsigned short* rA = vA ? xb + (size_t)(nA + dh * WW + dw) * CIN : zpad;
    const unsigned short* rB = vB ? xb + (size_t)(nB + dh * WW + dw) * CIN : zpad;
    const unsigned short* wa0 = w0p + (size_t)com * (9 * CIN) + tap * CIN;
    const unsigned short* wa1 = wa0 + (size_t)16 * (9 * CIN);

    // software pipeline: fragments for step ck are loaded at step ck-32
    v16bf a0  = load_a_frag(wa0, 9 * CIN, lane);
    v16bf a1  = load_a_frag(wa1, 9 * CIN, lane);
    v16bf fb0 = load_b_row(rA, lane);
    v16bf fb1 = load_b_row(rB, lane);
    for (int ck = 32; ck < CIN; ck += 32) {
      v16bf na0  = load_a_frag(wa0 + ck, 9 * CIN, lane);
      v16bf na1  = load_a_frag(wa1 + ck, 9 * CIN, lane);
      v16bf nb0  = load_b_row(rA + ck, lane);
      v16bf nb1  = load_b_row(rB + ck, lane);
      acc[0][0] = wmma_bf16(a0, fb0, acc[0][0]);
      acc[0][1] = wmma_bf16(a0, fb1, acc[0][1]);
      acc[1][0] = wmma_bf16(a1, fb0, acc[1][0]);
      acc[1][1] = wmma_bf16(a1, fb1, acc[1][1]);
      a0 = na0; a1 = na1; fb0 = nb0; fb1 = nb1;
    }
    acc[0][0] = wmma_bf16(a0, fb0, acc[0][0]);
    acc[0][1] = wmma_bf16(a0, fb1, acc[0][1]);
    acc[1][0] = wmma_bf16(a1, fb0, acc[1][0]);
    acc[1][1] = wmma_bf16(a1, fb1, acc[1][1]);
  }

#pragma unroll
  for (int mf = 0; mf < 2; ++mf)
#pragma unroll
    for (int nf = 0; nf < 2; ++nf) {
      const int n = nnb + nf * 16 + ln;
      float r[8];
#pragma unroll
      for (int i = 0; i < 8; ++i) {
        const int co = com + mf * 16 + i + 8 * hl;
        float v = acc[mf][nf][i] * s0[co] + t0[co];
        v = v > 0.f ? v : 0.f;
        r[i] = v;
        idn[((size_t)b * CR + co) * NN + n] = v;
      }
      unsigned* dn = (unsigned*)(idn_nc + ((size_t)b * NN + n) * CR +
                                 com + mf * 16 + 8 * hl);
#pragma unroll
      for (int j = 0; j < 4; ++j) dn[j] = pack_bf16x2(r[2 * j], r[2 * j + 1]);
    }
}

// ----------------------------- conv1: 1x1 + bias ---------------------------
__global__ void __launch_bounds__(256)
conv1_bias_k(const unsigned short* __restrict__ idn_nc,   // [B][N][CR]
             const unsigned short* __restrict__ w1p,      // [CR][CR]
             const float* __restrict__ bias1,
             unsigned short* __restrict__ xf_nc,          // [B][N][CR]
             unsigned short* __restrict__ xf_cn) {        // [B][CR][N]
  const int b = blockIdx.z;
  const int co0 = blockIdx.y * 64;
  const int n0  = blockIdx.x * 128;
  const int lane = threadIdx.x & 31, wave = threadIdx.x >> 5;
  const int com = co0 + (wave >> 2) * 32;
  const int nnb = n0 + (wave & 3) * 32;
  const int ln = lane & 15, hl = lane >> 4;

  v8f acc[2][2] = {};
  const unsigned short* A0 = w1p + (size_t)com * CR;
  const unsigned short* Bt = idn_nc + ((size_t)b * NN + nnb) * CR;
  for (int ck = 0; ck < CR; ck += 32) {
    v16bf a0 = load_a_frag(A0 + ck, CR, lane);
    v16bf a1 = load_a_frag(A0 + 16 * CR + ck, CR, lane);
    v16bf fb0 = load_b_frag(Bt + ck, CR, lane);
    v16bf fb1 = load_b_frag(Bt + 16 * CR + ck, CR, lane);
    acc[0][0] = wmma_bf16(a0, fb0, acc[0][0]);
    acc[0][1] = wmma_bf16(a0, fb1, acc[0][1]);
    acc[1][0] = wmma_bf16(a1, fb0, acc[1][0]);
    acc[1][1] = wmma_bf16(a1, fb1, acc[1][1]);
  }
#pragma unroll
  for (int mf = 0; mf < 2; ++mf)
#pragma unroll
    for (int nf = 0; nf < 2; ++nf) {
      const int n = nnb + nf * 16 + ln;
      float r[8];
#pragma unroll
      for (int i = 0; i < 8; ++i) {
        const int co = com + mf * 16 + i + 8 * hl;
        const float v = acc[mf][nf][i] + bias1[co];
        r[i] = v;
        xf_cn[((size_t)b * CR + co) * NN + n] = f32_bf16(v);
      }
      unsigned* dn = (unsigned*)(xf_nc + ((size_t)b * NN + n) * CR +
                                 com + mf * 16 + 8 * hl);
#pragma unroll
      for (int j = 0; j < 4; ++j) dn[j] = pack_bf16x2(r[2 * j], r[2 * j + 1]);
    }
}

// ---------------- EM: z = softmax(xf^T mu), S[k] += colsum ----------------
// grid (144, 1, B), block 256. WG computes z tile [64 n][64 k].
__global__ void __launch_bounds__(256)
z_softmax_k(const unsigned short* __restrict__ xf_nc,   // [B][N][CR]
            const unsigned short* __restrict__ mu_kc,   // [B][KK][CR]
            unsigned short* __restrict__ z_nk,          // [B][N][KK]
            unsigned short* __restrict__ z_kn,          // [B][KK][N]
            float* __restrict__ S) {                    // [B][KK]
  __shared__ float zt[64][65];
  const int b = blockIdx.z, n0 = blockIdx.x * 64;
  const int lane = threadIdx.x & 31, wave = threadIdx.x >> 5;
  const int wm = wave >> 1, wn = wave & 1;
  const int ln = lane & 15, hl = lane >> 4;

  v8f acc[2] = {};
  const unsigned short* A0 = xf_nc + ((size_t)b * NN + n0 + wm * 16) * CR;
  const unsigned short* B0 = mu_kc + ((size_t)b * KK + wn * 32) * CR;
  for (int ck = 0; ck < CR; ck += 32) {
    v16bf a  = load_a_frag(A0 + ck, CR, lane);
    v16bf g0 = load_b_frag(B0 + ck, CR, lane);
    v16bf g1 = load_b_frag(B0 + 16 * CR + ck, CR, lane);
    acc[0] = wmma_bf16(a, g0, acc[0]);
    acc[1] = wmma_bf16(a, g1, acc[1]);
  }
#pragma unroll
  for (int f = 0; f < 2; ++f)
#pragma unroll
    for (int i = 0; i < 8; ++i)
      zt[wm * 16 + i + 8 * hl][wn * 32 + f * 16 + ln] = acc[f][i];
  __syncthreads();

  if (threadIdx.x < 64) {               // softmax along k for one pixel row
    const int r = threadIdx.x;
    float mx = -3.4e38f;
    for (int k = 0; k < KK; ++k) mx = fmaxf(mx, zt[r][k]);
    float s = 0.f;
    for (int k = 0; k < KK; ++k) { float e = __expf(zt[r][k] - mx); zt[r][k] = e; s += e; }
    const float inv = 1.f / s;
    for (int k = 0; k < KK; ++k) zt[r][k] *= inv;
  }
  __syncthreads();

  if (threadIdx.x < 64) {               // partial column sums -> global S
    const int k = threadIdx.x;
    float cs = 0.f;
    for (int r = 0; r < 64; ++r) cs += zt[r][k];
    atomicAdd(S + b * KK + k, cs);
  }
  for (int e = threadIdx.x; e < 64 * 64; e += 256) {
    const int r = e >> 6, k = e & 63;
    const unsigned short v = f32_bf16(zt[r][k]);
    z_nk[((size_t)b * NN + n0 + r) * KK + k] = v;
    z_kn[((size_t)b * KK + k) * NN + n0 + r] = v;
  }
}

// ------------- EM: mu_raw[c,k] += xf @ z (split-K over pixels) -------------
// grid (8 splits, 4 c-tiles, B), block 256 (8 waves: 4(M) x 2(N-halves)).
__global__ void __launch_bounds__(256)
mu_update_k(const unsigned short* __restrict__ xf_cn,   // [B][CR][N]
            const unsigned short* __restrict__ z_kn,    // [B][KK][N]
            float* __restrict__ mu_raw) {               // [B][CR][KK]
  const int b = blockIdx.z;
  const int c0 = blockIdx.y * 64;
  const int ns = blockIdx.x * (NN / 8);                 // 1152 pixels per split
  const int lane = threadIdx.x & 31, wave = threadIdx.x >> 5;
  const int cm = c0 + (wave >> 1) * 16;
  const int k0 = (wave & 1) * 32;
  const int ln = lane & 15, hl = lane >> 4;

  v8f acc[2] = {};
  const unsigned short* A0 = xf_cn + ((size_t)b * CR + cm) * NN + ns;
  const unsigned short* B0 = z_kn + ((size_t)b * KK + k0) * NN + ns;
  for (int nk = 0; nk < NN / 8; nk += 32) {
    v16bf a  = load_a_frag(A0 + nk, NN, lane);
    v16bf g0 = load_b_frag(B0 + nk, NN, lane);
    v16bf g1 = load_b_frag(B0 + 16 * NN + nk, NN, lane);
    acc[0] = wmma_bf16(a, g0, acc[0]);
    acc[1] = wmma_bf16(a, g1, acc[1]);
  }
#pragma unroll
  for (int f = 0; f < 2; ++f)
#pragma unroll
    for (int i = 0; i < 8; ++i) {
      const int c = cm + i + 8 * hl;
      const int k = k0 + f * 16 + ln;
      atomicAdd(mu_raw + ((size_t)b * CR + c) * KK + k, acc[f][i]);
    }
}

// -------- EM: mu = l2norm(mu_raw / (eps+S)); write both bf16 layouts -------
// grid 256 blocks (b*k), block 256 threads (c).
__global__ void __launch_bounds__(256)
mu_norm_k(const float* __restrict__ mu_raw, const float* __restrict__ S,
          unsigned short* __restrict__ mu_kc, unsigned short* __restrict__ mu_ck,
          float* __restrict__ mu_out, int write_out) {
  __shared__ float red[256];
  const int b = blockIdx.x >> 6, k = blockIdx.x & 63, c = threadIdx.x;
  const float v = mu_raw[((size_t)b * CR + c) * KK + k] / (EPS_EM + S[b * KK + k]);
  red[c] = v * v;
  __syncthreads();
  for (int s = 128; s > 0; s >>= 1) {
    if (c < s) red[c] += red[c + s];
    __syncthreads();
  }
  const float o = v / (EPS_EM + sqrtf(red[0]));
  mu_kc[((size_t)b * KK + k) * CR + c] = f32_bf16(o);
  mu_ck[((size_t)b * CR + c) * KK + k] = f32_bf16(o);
  if (write_out) mu_out[((size_t)b * CR + c) * KK + k] = o;
}

// --------------------- reconstruct: x = relu(mu @ z^T) ---------------------
__global__ void __launch_bounds__(256)
recon_relu_k(const unsigned short* __restrict__ mu_ck,  // [B][CR][KK]
             const unsigned short* __restrict__ z_nk,   // [B][N][KK]
             unsigned short* __restrict__ xr_nc) {      // [B][N][CR]
  const int b = blockIdx.z;
  const int c0 = blockIdx.y * 64;
  const int n0 = blockIdx.x * 128;
  const int lane = threadIdx.x & 31, wave = threadIdx.x >> 5;
  const int cm = c0 + (wave >> 2) * 32;
  const int nnb = n0 + (wave & 3) * 32;
  const int ln = lane & 15, hl = lane >> 4;

  v8f acc[2][2] = {};
  const unsigned short* A0 = mu_ck + ((size_t)b * CR + cm) * KK;
  const unsigned short* Bt = z_nk + ((size_t)b * NN + nnb) * KK;
#pragma unroll
  for (int kk = 0; kk < KK; kk += 32) {
    v16bf a0 = load_a_frag(A0 + kk, KK, lane);
    v16bf a1 = load_a_frag(A0 + 16 * KK + kk, KK, lane);
    v16bf g0 = load_b_frag(Bt + kk, KK, lane);
    v16bf g1 = load_b_frag(Bt + 16 * KK + kk, KK, lane);
    acc[0][0] = wmma_bf16(a0, g0, acc[0][0]);
    acc[0][1] = wmma_bf16(a0, g1, acc[0][1]);
    acc[1][0] = wmma_bf16(a1, g0, acc[1][0]);
    acc[1][1] = wmma_bf16(a1, g1, acc[1][1]);
  }
#pragma unroll
  for (int mf = 0; mf < 2; ++mf)
#pragma unroll
    for (int nf = 0; nf < 2; ++nf) {
      const int n = nnb + nf * 16 + ln;
      float r[8];
#pragma unroll
      for (int i = 0; i < 8; ++i) {
        float v = acc[mf][nf][i];
        r[i] = v > 0.f ? v : 0.f;
      }
      unsigned* dn = (unsigned*)(xr_nc + ((size_t)b * NN + n) * CR +
                                 cm + mf * 16 + 8 * hl);
#pragma unroll
      for (int j = 0; j < 4; ++j) dn[j] = pack_bf16x2(r[2 * j], r[2 * j + 1]);
    }
}

// ----------------- conv2: 1x1 -> BN -> + idn -> ReLU -> out ----------------
__global__ void __launch_bounds__(256)
conv2_bn_add_relu_k(const unsigned short* __restrict__ xr_nc, // [B][N][CR]
                    const unsigned short* __restrict__ w2p,   // [CR][CR]
                    const float* __restrict__ s2, const float* __restrict__ t2,
                    const float* __restrict__ idn,            // [B][CR][N]
                    float* __restrict__ out) {                // [B][CR][N]
  const int b = blockIdx.z;
  const int co0 = blockIdx.y * 64;
  const int n0  = blockIdx.x * 128;
  const int lane = threadIdx.x & 31, wave = threadIdx.x >> 5;
  const int com = co0 + (wave >> 2) * 32;
  const int nnb = n0 + (wave & 3) * 32;
  const int ln = lane & 15, hl = lane >> 4;

  v8f acc[2][2] = {};
  const unsigned short* A0 = w2p + (size_t)com * CR;
  const unsigned short* Bt = xr_nc + ((size_t)b * NN + nnb) * CR;
  for (int ck = 0; ck < CR; ck += 32) {
    v16bf a0 = load_a_frag(A0 + ck, CR, lane);
    v16bf a1 = load_a_frag(A0 + 16 * CR + ck, CR, lane);
    v16bf g0 = load_b_frag(Bt + ck, CR, lane);
    v16bf g1 = load_b_frag(Bt + 16 * CR + ck, CR, lane);
    acc[0][0] = wmma_bf16(a0, g0, acc[0][0]);
    acc[0][1] = wmma_bf16(a0, g1, acc[0][1]);
    acc[1][0] = wmma_bf16(a1, g0, acc[1][0]);
    acc[1][1] = wmma_bf16(a1, g1, acc[1][1]);
  }
#pragma unroll
  for (int mf = 0; mf < 2; ++mf)
#pragma unroll
    for (int nf = 0; nf < 2; ++nf)
#pragma unroll
      for (int i = 0; i < 8; ++i) {
        const int co = com + mf * 16 + i + 8 * hl;
        const int n  = nnb + nf * 16 + ln;
        const size_t idx = ((size_t)b * CR + co) * NN + n;
        float v = acc[mf][nf][i] * s2[co] + t2[co] + idn[idx];
        v = v > 0.f ? v : 0.f;
        out[idx] = v;
      }
}

// ------------------------------- host launch -------------------------------

extern "C" void kernel_launch(void* const* d_in, const int* in_sizes, int n_in,
                              void* d_out, int out_size, void* d_ws, size_t ws_size,
                              hipStream_t stream) {
  (void)in_sizes; (void)n_in; (void)out_size; (void)ws_size;
  const float* x   = (const float*)d_in[0];
  const float* w0  = (const float*)d_in[1];
  const float* g0  = (const float*)d_in[2];
  const float* be0 = (const float*)d_in[3];
  const float* m0  = (const float*)d_in[4];
  const float* v0  = (const float*)d_in[5];
  const float* w1  = (const float*)d_in[6];
  const float* b1  = (const float*)d_in[7];
  const float* w2  = (const float*)d_in[8];
  const float* g2  = (const float*)d_in[9];
  const float* be2 = (const float*)d_in[10];
  const float* m2  = (const float*)d_in[11];
  const float* v2  = (const float*)d_in[12];
  const float* mu  = (const float*)d_in[13];
  float* out = (float*)d_out;
  float* mu_out = out + (size_t)B_ * CR * NN;

  char* ws = (char*)d_ws;
  size_t off = 0;
  auto take = [&](size_t bytes) -> char* {
    char* p = ws + off;
    off += (bytes + 255) & ~(size_t)255;
    return p;
  };
  // ~203 MB of workspace
  unsigned short* xnc    = (unsigned short*)take((size_t)B_ * NN * CIN * 2);
  unsigned short* w0p    = (unsigned short*)take((size_t)CR * 9 * CIN * 2);
  unsigned short* w1p    = (unsigned short*)take((size_t)CR * CR * 2);
  unsigned short* w2p    = (unsigned short*)take((size_t)CR * CR * 2);
  unsigned short* zpad   = (unsigned short*)take((size_t)(CIN + 64) * 2);
  float*          s0     = (float*)take(CR * 4);
  float*          t0     = (float*)take(CR * 4);
  float*          s2     = (float*)take(CR * 4);
  float*          t2     = (float*)take(CR * 4);
  float*          idn    = (float*)take((size_t)B_ * CR * NN * 4);
  unsigned short* idn_nc = (unsigned short*)take((size_t)B_ * NN * CR * 2);
  unsigned short* xf_nc  = (unsigned short*)take((size_t)B_ * NN * CR * 2);
  unsigned short* xf_cn  = (unsigned short*)take((size_t)B_ * CR * NN * 2);
  unsigned short* mu_kc  = (unsigned short*)take((size_t)B_ * KK * CR * 2);
  unsigned short* mu_ck  = (unsigned short*)take((size_t)B_ * CR * KK * 2);
  unsigned short* z_nk   = (unsigned short*)take((size_t)B_ * NN * KK * 2);
  unsigned short* z_kn   = (unsigned short*)take((size_t)B_ * KK * NN * 2);
  float*          Ssum   = (float*)take((size_t)B_ * KK * 4);
  float*          mu_raw = (float*)take((size_t)B_ * CR * KK * 4);
  unsigned short* xr_nc  = (unsigned short*)take((size_t)B_ * NN * CR * 2);

  // ---- prep / packing ----
  prep_bn_k<<<1, 256, 0, stream>>>(g0, be0, m0, v0, g2, be2, m2, v2,
                                   s0, t0, s2, t2, zpad);
  pack_w0_k<<<(CR * 9 * CIN) / 256, 256, 0, stream>>>(w0, w0p);
  pack_bf16_k<<<(CR * CR) / 256, 256, 0, stream>>>(w1, w1p, CR * CR);
  pack_bf16_k<<<(CR * CR) / 256, 256, 0, stream>>>(w2, w2p, CR * CR);
  x_transpose_k<<<dim3(NN / 64, CIN / 64, B_), 256, 0, stream>>>(x, xnc);

  // ---- conv0 (dominant GEMM) + BN + ReLU ----
  conv0_bn_relu_k<<<dim3(NN / 128, CR / 64, B_), 256, 0, stream>>>(
      xnc, w0p, zpad, s0, t0, idn, idn_nc);

  // ---- conv1 + bias ----
  conv1_bias_k<<<dim3(NN / 128, CR / 64, B_), 256, 0, stream>>>(
      idn_nc, w1p, b1, xf_nc, xf_cn);

  // ---- EM iterations ----
  mu_init_k<<<(B_ * CR * KK) / 256, 256, 0, stream>>>(mu, mu_kc, mu_ck);
  for (int s = 0; s < 3; ++s) {
    zero_s_mu_k<<<(B_ * CR * KK) / 256, 256, 0, stream>>>(Ssum, mu_raw);
    z_softmax_k<<<dim3(NN / 64, 1, B_), 256, 0, stream>>>(xf_nc, mu_kc, z_nk, z_kn, Ssum);
    mu_update_k<<<dim3(8, CR / 64, B_), 256, 0, stream>>>(xf_cn, z_kn, mu_raw);
    mu_norm_k<<<B_ * KK, 256, 0, stream>>>(mu_raw, Ssum, mu_kc, mu_ck, mu_out,
                                           (s == 2) ? 1 : 0);
  }

  // ---- reconstruct + ReLU ----
  recon_relu_k<<<dim3(NN / 128, CR / 64, B_), 256, 0, stream>>>(mu_ck, z_nk, xr_nc);

  // ---- conv2 + BN + residual + ReLU -> out ----
  conv2_bn_add_relu_k<<<dim3(NN / 128, CR / 64, B_), 256, 0, stream>>>(
      xr_nc, w2p, s2, t2, idn, out);
}
